// RNNClassifier_70592082477411
// MI455X (gfx1250) — matrix operand order repaired and checked by reading, not compiled, not run on hardware
//
#include <hip/hip_runtime.h>
#include <hip/hip_bf16.h>
#include <math.h>

// ---------------------------------------------------------------------------
// Types for CDNA5 WMMA (wave32): bf16 A/B fragments (16 elems/lane), f32 accum.
// ---------------------------------------------------------------------------
typedef __bf16 bf16_t;
typedef __attribute__((ext_vector_type(16))) __bf16 v16bf;
typedef __attribute__((ext_vector_type(8)))  __bf16 v8bf;
typedef __attribute__((ext_vector_type(8)))  float  v8f;

#define DEV __device__ __forceinline__

union Frag16 { v16bf v; v8bf h[2]; };

DEV unsigned short bf16bits(float f) {
  unsigned u = __float_as_uint(f);
  u += 0x7FFFu + ((u >> 16) & 1u);     // round-to-nearest-even
  return (unsigned short)(u >> 16);
}
DEV float sigf(float x) { return 1.0f / (1.0f + __expf(-x)); }

// A fragment: 16x32 bf16 (MxK). ISA layout: lanes0-15 rows M=0..15, K=kb..kb+7
// in v[0..7] and K=kb+16..kb+23 in v[8..15]; kb = 0 (lanes<16) or 8 (lanes>=16).
DEV v16bf load_a_frag(const bf16_t* __restrict__ A, int lda, int row0, int k0) {
  const int lane = threadIdx.x & 31;
  const int row  = row0 + (lane & 15);
  const int kb   = k0 + ((lane >> 4) << 3);
  const bf16_t* p = A + (size_t)row * lda + kb;
  Frag16 f;
  f.h[0] = *(const v8bf*)(p);
  f.h[1] = *(const v8bf*)(p + 16);
  return f.v;
}

// B fragment: 32x16 bf16 (KxN). Lane n holds column N=n&15, K=kb..kb+7 in
// v[0..7] and K=kb+16..kb+23 in v[8..15], kb = 0 (lanes<16) or 8 (lanes>=16).
// With W stored [N][K] row-major this is two 16B runs per lane.
DEV v16bf load_b_frag(const bf16_t* __restrict__ W, int ldw, int col0, int k0) {
  const int lane = threadIdx.x & 31;
  const int col  = col0 + (lane & 15);
  const int kb   = k0 + ((lane >> 4) << 3);
  const bf16_t* p = W + (size_t)col * ldw + kb;
  Frag16 f;
  f.h[0] = *(const v8bf*)(p);
  f.h[1] = *(const v8bf*)(p + 16);
  return f.v;
}

DEV v8f wmma_bf16(v16bf a, v16bf b, v8f c) {
  return __builtin_amdgcn_wmma_f32_16x16x32_bf16(false, a, false, b, (short)0, c,
                                                 false, false);
}

// ---------------------------------------------------------------------------
// Elementwise helpers
// ---------------------------------------------------------------------------
__global__ void cvt_pad_bf16_kernel(const float* __restrict__ src,
                                    unsigned short* __restrict__ dst,
                                    int rows, int ks, int kd) {
  size_t i = (size_t)blockIdx.x * blockDim.x + threadIdx.x;
  size_t total = (size_t)rows * kd;
  if (i >= total) return;
  int r = (int)(i / kd), k = (int)(i % kd);
  dst[i] = (k < ks) ? bf16bits(src[(size_t)r * ks + k]) : (unsigned short)0;
}

__global__ void combine_bias_kernel(const float* __restrict__ a,
                                    const float* __restrict__ b,
                                    float* __restrict__ dst, int n) {
  int i = blockIdx.x * blockDim.x + threadIdx.x;
  if (i < n) dst[i] = a[i] + b[i];
}

// Embedding gather -> bf16 with K-padding (E=300 -> 320).
__global__ void embed_kernel(const int* __restrict__ tok,
                             const float* __restrict__ emb,
                             unsigned short* __restrict__ dst,
                             int BT, int E, int KP) {
  size_t i = (size_t)blockIdx.x * blockDim.x + threadIdx.x;
  size_t total = (size_t)BT * KP;
  if (i >= total) return;
  int r = (int)(i / KP), k = (int)(i % KP);
  if (k < E) {
    int t = tok[r];
    dst[i] = bf16bits(emb[(size_t)t * E + k]);
  } else {
    dst[i] = 0;
  }
}

__global__ void init_rec_kernel(unsigned short* __restrict__ hbuf,
                                unsigned* __restrict__ cnt) {
  size_t i = (size_t)blockIdx.x * blockDim.x + threadIdx.x;
  if (i < (size_t)2 * 64 * 512) hbuf[i] = 0;
  if (i == 0) *cnt = 0u;
}

// ---------------------------------------------------------------------------
// Input projection GEMM:  XP[m][n] = sum_k X[m][k]*W[n][k] + bias[n]
// X: [M x K] bf16, W: [2048 x K] bf16 (row-major = already transposed for B),
// XP: [M x 2048] f32.  Block = 256 thr = 8 waves (2 M x 4 N), wave tile 32x32.
// Block tile 64(M) x 128(N). M,N,K all divide exactly.
// ---------------------------------------------------------------------------
__global__ __launch_bounds__(256)
void gemm_xp_kernel(const bf16_t* __restrict__ X, const bf16_t* __restrict__ W,
                    const float* __restrict__ bias, float* __restrict__ XP,
                    int K) {
  const int N = 2048;
  const int wave = threadIdx.x >> 5;
  const int wm = wave & 1;
  const int wn = wave >> 1;
  const int m_base = blockIdx.y * 64 + wm * 32;
  const int n_base = blockIdx.x * 128 + wn * 32;

  v8f acc00 = {}, acc01 = {}, acc10 = {}, acc11 = {};
  for (int k0 = 0; k0 < K; k0 += 32) {
    v16bf a0 = load_a_frag(X, K, m_base, k0);
    v16bf a1 = load_a_frag(X, K, m_base + 16, k0);
    v16bf b0 = load_b_frag(W, K, n_base, k0);
    v16bf b1 = load_b_frag(W, K, n_base + 16, k0);
    acc00 = wmma_bf16(a0, b0, acc00);
    acc01 = wmma_bf16(a0, b1, acc01);
    acc10 = wmma_bf16(a1, b0, acc10);
    acc11 = wmma_bf16(a1, b1, acc11);
  }

  const int lane = threadIdx.x & 31;
  const int nlo = lane & 15;
  const int mhi = (lane >> 4) << 3;
  v8f accs[2][2] = {{acc00, acc01}, {acc10, acc11}};
#pragma unroll
  for (int ti = 0; ti < 2; ++ti) {
#pragma unroll
    for (int tj = 0; tj < 2; ++tj) {
      int n = n_base + tj * 16 + nlo;
      float bv = bias[n];
#pragma unroll
      for (int r = 0; r < 8; ++r) {
        int m = m_base + ti * 16 + mhi + r;
        XP[(size_t)m * N + n] = accs[ti][tj][r] + bv;
      }
    }
  }
}

// ---------------------------------------------------------------------------
// Persistent recurrent LSTM kernel (one direction of one layer).
// grid = 32 WGs x 256 threads. WG g owns hidden units j0=g*16 .. j0+15
// (all four gates for those units). Per step:
//   G[64 x 64cols] = h[64x512](bf16) @ Whh_cols^T  (WMMA)  + XP[t]
// then the i/f/g/o combine + cell update in LDS, h written bf16 to a
// double-buffered global h, device-wide barrier via monotonic atomic counter.
// ---------------------------------------------------------------------------
__global__ __launch_bounds__(256)
void lstm_rec_kernel(const float* __restrict__ XP,      // [B,T,2048] (+bias)
                     const bf16_t* __restrict__ Whh,    // [2048,512] bf16
                     float* __restrict__ Y,             // [B,T,1024]
                     bf16_t* __restrict__ hbuf,         // [2,64,512] bf16
                     unsigned* __restrict__ bar,
                     int T, int reverse, int dirOff) {
  const int H = 512, Bsz = 64, G = 2048;
  __shared__ float g_lds[4 * 64 * 16];   // [gate][b][jj]
  __shared__ float c_lds[64 * 16];       // cell state for this WG's slice

  const int j0 = blockIdx.x * 16;
  for (int i = threadIdx.x; i < 64 * 16; i += 256) c_lds[i] = 0.0f;
  __syncthreads();

  const int wave = threadIdx.x >> 5;
  const int lane = threadIdx.x & 31;
  const int wm = wave & 1;    // 2 waves in M (batch)
  const int wn = wave >> 1;   // 4 waves = 4 gates
  const unsigned nwg = gridDim.x;

  for (int t = 0; t < T; ++t) {
    const int tt = reverse ? (T - 1 - t) : t;
    const bf16_t* hcur = hbuf + (size_t)(t & 1) * Bsz * H;
    bf16_t* hnxt = hbuf + (size_t)((t + 1) & 1) * Bsz * H;

    v8f acc0 = {}, acc1 = {};
    const int ncol = wn * 512 + j0;
    for (int k0 = 0; k0 < H; k0 += 32) {
      v16bf a0 = load_a_frag(hcur, H, wm * 32, k0);
      v16bf a1 = load_a_frag(hcur, H, wm * 32 + 16, k0);
      v16bf bb = load_b_frag(Whh, H, ncol, k0);
      acc0 = wmma_bf16(a0, bb, acc0);
      acc1 = wmma_bf16(a1, bb, acc1);
    }

    // add input projection, stage gates in LDS
    const int jj = lane & 15;
    const int mhi = (lane >> 4) << 3;
    v8f accs[2] = {acc0, acc1};
#pragma unroll
    for (int a = 0; a < 2; ++a) {
#pragma unroll
      for (int r = 0; r < 8; ++r) {
        int bi = wm * 32 + a * 16 + mhi + r;
        int n = wn * 512 + j0 + jj;
        float g = accs[a][r] + XP[((size_t)bi * T + tt) * G + n];
        g_lds[(wn * 64 + bi) * 16 + jj] = g;
      }
    }
    __syncthreads();

    // elementwise gate combine + state update for this WG's 64x16 slice
    for (int idx = threadIdx.x; idx < 64 * 16; idx += 256) {
      int bi = idx >> 4, j = idx & 15;
      float gi = g_lds[(0 * 64 + bi) * 16 + j];
      float gf = g_lds[(1 * 64 + bi) * 16 + j];
      float gg = g_lds[(2 * 64 + bi) * 16 + j];
      float go = g_lds[(3 * 64 + bi) * 16 + j];
      float c = sigf(gf) * c_lds[idx] + sigf(gi) * tanhf(gg);
      float h = sigf(go) * tanhf(c);
      c_lds[idx] = c;
      ((unsigned short*)hnxt)[bi * H + j0 + j] = bf16bits(h);
      Y[((size_t)bi * T + tt) * 1024 + dirOff + j0 + j] = h;
    }

    // device-wide barrier (monotonic counter; WGs trivially co-resident)
    __threadfence();
    __syncthreads();
    if (threadIdx.x == 0) {
      atomicAdd(bar, 1u);
      const unsigned target = nwg * (unsigned)(t + 1);
      while (atomicAdd(bar, 0u) < target) { __builtin_amdgcn_s_sleep(1); }
    }
    __syncthreads();
    __threadfence();
  }
}

// ---------------------------------------------------------------------------
// Final classifier: feat = [Y1[:,T-1,:512], Y1[:,0,512:]], logits = feat@W^T+b,
// log_softmax over C=10. One block.
// ---------------------------------------------------------------------------
__global__ __launch_bounds__(256)
void classifier_kernel(const float* __restrict__ Y1,
                       const float* __restrict__ OW,
                       const float* __restrict__ OB,
                       float* __restrict__ out, int T) {
  __shared__ float lg[64 * 10];
  for (int idx = threadIdx.x; idx < 640; idx += 256) {
    int b = idx / 10, c = idx % 10;
    const float* fwd = Y1 + ((size_t)b * T + (T - 1)) * 1024;      // [:512]
    const float* bwd = Y1 + ((size_t)b * T + 0) * 1024 + 512;      // [512:]
    float s = OB[c];
    for (int j = 0; j < 512; ++j) s += fwd[j] * OW[c * 1024 + j];
    for (int j = 0; j < 512; ++j) s += bwd[j] * OW[c * 1024 + 512 + j];
    lg[idx] = s;
  }
  __syncthreads();
  if (threadIdx.x < 64) {
    int b = threadIdx.x;
    float mx = -1e30f;
    for (int c = 0; c < 10; ++c) mx = fmaxf(mx, lg[b * 10 + c]);
    float se = 0.0f;
    for (int c = 0; c < 10; ++c) se += __expf(lg[b * 10 + c] - mx);
    float lse = mx + __logf(se);
    for (int c = 0; c < 10; ++c) out[b * 10 + c] = lg[b * 10 + c] - lse;
  }
}

// ---------------------------------------------------------------------------
// Host orchestration
// ---------------------------------------------------------------------------
extern "C" void kernel_launch(void* const* d_in, const int* in_sizes, int n_in,
                              void* d_out, int out_size, void* d_ws, size_t ws_size,
                              hipStream_t stream) {
  (void)in_sizes; (void)n_in; (void)out_size; (void)ws_size;
  const int B = 64, T = 512, E = 300, EP = 320, H = 512, G = 2048;
  const int BT = B * T;

  const int*   tok  = (const int*)d_in[0];
  const float* emb  = (const float*)d_in[1];
  // index [layer*2 + dir]: 0=l0f 1=l0b 2=l1f 3=l1b
  const float* wih[4] = {(const float*)d_in[2],  (const float*)d_in[6],
                         (const float*)d_in[10], (const float*)d_in[14]};
  const float* whh[4] = {(const float*)d_in[3],  (const float*)d_in[7],
                         (const float*)d_in[11], (const float*)d_in[15]};
  const float* bih[4] = {(const float*)d_in[4],  (const float*)d_in[8],
                         (const float*)d_in[12], (const float*)d_in[16]};
  const float* bhh[4] = {(const float*)d_in[5],  (const float*)d_in[9],
                         (const float*)d_in[13], (const float*)d_in[17]};
  const float* outw = (const float*)d_in[18];
  const float* outb = (const float*)d_in[19];
  float* out = (float*)d_out;

  // ---- workspace layout (256B-aligned slices) ----
  char* base = (char*)d_ws;
  size_t off = 0;
  auto alloc = [&](size_t bytes) -> char* {
    char* p = base + off;
    off = (off + bytes + 255) & ~(size_t)255;
    return p;
  };
  unsigned*       cnt   = (unsigned*)alloc(256);
  unsigned short* hbuf  = (unsigned short*)alloc((size_t)2 * B * H * 2);
  unsigned short* x0b   = (unsigned short*)alloc((size_t)BT * EP * 2);
  unsigned short* wihb[4];
  wihb[0] = (unsigned short*)alloc((size_t)G * EP * 2);
  wihb[1] = (unsigned short*)alloc((size_t)G * EP * 2);
  wihb[2] = (unsigned short*)alloc((size_t)G * 1024 * 2);
  wihb[3] = (unsigned short*)alloc((size_t)G * 1024 * 2);
  unsigned short* whhb[4];
  for (int i = 0; i < 4; ++i) whhb[i] = (unsigned short*)alloc((size_t)G * H * 2);
  float* biasc[4];
  for (int i = 0; i < 4; ++i) biasc[i] = (float*)alloc((size_t)G * 4);
  float*          XP   = (float*)alloc((size_t)BT * G * 4);
  float*          Y0   = (float*)alloc((size_t)BT * 1024 * 4);
  float*          Y1   = (float*)alloc((size_t)BT * 1024 * 4);
  unsigned short* y0b  = (unsigned short*)alloc((size_t)BT * 1024 * 2);

  auto blocks = [](size_t total) { return (unsigned)((total + 255) / 256); };

  // ---- weight / bias conversion ----
  cvt_pad_bf16_kernel<<<blocks((size_t)G * EP), 256, 0, stream>>>(wih[0], wihb[0], G, E, EP);
  cvt_pad_bf16_kernel<<<blocks((size_t)G * EP), 256, 0, stream>>>(wih[1], wihb[1], G, E, EP);
  cvt_pad_bf16_kernel<<<blocks((size_t)G * 1024), 256, 0, stream>>>(wih[2], wihb[2], G, 1024, 1024);
  cvt_pad_bf16_kernel<<<blocks((size_t)G * 1024), 256, 0, stream>>>(wih[3], wihb[3], G, 1024, 1024);
  for (int i = 0; i < 4; ++i) {
    cvt_pad_bf16_kernel<<<blocks((size_t)G * H), 256, 0, stream>>>(whh[i], whhb[i], G, H, H);
    combine_bias_kernel<<<blocks(G), 256, 0, stream>>>(bih[i], bhh[i], biasc[i], G);
  }

  // ---- embedding gather (f32 -> bf16, pad 300 -> 320) ----
  embed_kernel<<<blocks((size_t)BT * EP), 256, 0, stream>>>(tok, emb, x0b, BT, E, EP);

  // ---- layer 0 (K=320) then layer 1 (K=1024) ----
  for (int layer = 0; layer < 2; ++layer) {
    const bf16_t* Xin = (const bf16_t*)(layer == 0 ? x0b : y0b);
    const int K = (layer == 0) ? EP : 1024;
    float* Y = (layer == 0) ? Y0 : Y1;
    for (int dir = 0; dir < 2; ++dir) {
      const int wi = layer * 2 + dir;
      dim3 ggrid(2048 / 128, BT / 64);
      gemm_xp_kernel<<<ggrid, 256, 0, stream>>>(Xin, (const bf16_t*)wihb[wi],
                                                biasc[wi], XP, K);
      init_rec_kernel<<<blocks((size_t)2 * B * H), 256, 0, stream>>>(hbuf, cnt);
      lstm_rec_kernel<<<H / 16, 256, 0, stream>>>(XP, (const bf16_t*)whhb[wi],
                                                  Y, (bf16_t*)hbuf, cnt,
                                                  T, dir, dir * 512);
    }
    if (layer == 0) {
      cvt_pad_bf16_kernel<<<blocks((size_t)BT * 1024), 256, 0, stream>>>(
          Y0, y0b, BT, 1024, 1024);
    }
  }

  // ---- classifier + log_softmax ----
  classifier_kernel<<<1, 256, 0, stream>>>(Y1, outw, outb, out, T);
}
